// QE_VTDD_43645457662008
// MI455X (gfx1250) — matrix operand order repaired and checked by reading, not compiled, not run on hardware
//
#include <hip/hip_runtime.h>
#include <hip/hip_bf16.h>

// ---------------------------------------------------------------------------
// Input index map (setup_inputs() dict insertion order, depth-first):
//   0            x                     (32,3,224,224) f32
//   1..5         cnn.conv1             w,gamma,beta,mean,var
//   6..25        layer1: blk0{conv1,conv2}, blk1{conv1,conv2}   (5 each)
//   26..50       layer2: blk0{conv1,conv2,down}, blk1{conv1,conv2}
//   51..75       layer3: same shape as layer2
//   76..100      layer4: same shape as layer2
//   101 fcr_w1 (64,512)  102 fcr_b1 (64)  103 fcr_w2 (4,64)  104 fcr_b2 (4)
//   105 qweights (3,4,3) 106 fc_w (2,4)   107 fc_b (2)
// Output: (32,2) f32 = 64 floats.
// ---------------------------------------------------------------------------

typedef _Float16 h16;
typedef __attribute__((ext_vector_type(16))) _Float16 v16h;
typedef __attribute__((ext_vector_type(8)))  float    v8f;

union V16 { v16h v; uint4 q[2]; };

// ---------------------------------------------------------------------------
// prep: fp32 OIHW weights -> fp16 [Cout][ky][kx][Cin], K padded to mult of 32
// ---------------------------------------------------------------------------
__global__ void prep_w_kernel(const float* __restrict__ w, h16* __restrict__ W16,
                              int Cout, int Cin, int kh, int kw, int Kpad) {
    int idx = blockIdx.x * blockDim.x + threadIdx.x;
    if (idx >= Cout * Kpad) return;
    int co = idx / Kpad, k = idx % Kpad;
    int Ktot = Cin * kh * kw;
    float v = 0.f;
    if (k < Ktot) {
        int ci = k % Cin; int r = k / Cin;
        int kx = r % kw;  int ky = r / kw;
        v = w[((long)(co * Cin + ci) * kh + ky) * kw + kx];
    }
    W16[idx] = (h16)v;
}

// BN fold: sb[0..C) = scale, sb[C..2C) = bias
__global__ void prep_bn_kernel(const float* __restrict__ g, const float* __restrict__ be,
                               const float* __restrict__ mn, const float* __restrict__ vr,
                               float* __restrict__ sb, int C) {
    int i = blockIdx.x * blockDim.x + threadIdx.x;
    if (i >= C) return;
    float s = g[i] * rsqrtf(vr[i] + 1e-5f);
    sb[i] = s;
    sb[C + i] = be[i] - mn[i] * s;
}

// NCHW f32 -> NHWC f16
__global__ void nchw_to_nhwc_kernel(const float* __restrict__ X, h16* __restrict__ Y,
                                    int C, int H, int W) {
    long idx = (long)blockIdx.x * blockDim.x + threadIdx.x;
    long total = 32L * H * W * C;
    if (idx >= total) return;
    int c = (int)(idx % C); long r = idx / C;
    int x = (int)(r % W); r /= W;
    int y = (int)(r % H); int b = (int)(r / H);
    Y[idx] = (h16)X[(((long)b * C + c) * H + y) * W + x];
}

// ---------------------------------------------------------------------------
// Fast path: Cin % 64 == 0 (all non-stem convs).  Implicit GEMM.
// Block tile M=64 x N=64, K-step 64.  8 waves; each wave owns a 32x16 C tile
// and issues 4 WMMAs per step (2 M-subtiles x 2 K-chunks, shared B fragment).
// Double-buffered LDS (2 x 8KB A + 2 x 8KB B): ONE barrier per K-step; the
// next tile's global b128 loads issue before the current tile's WMMAs.
// No div/mod in the loop: (ky,kx,ci0) advanced incrementally.
// ---------------------------------------------------------------------------
__global__ __launch_bounds__(256)
void conv_wmma_aligned_kernel(const h16* __restrict__ X, const h16* __restrict__ W,
                              const float* __restrict__ sb, const h16* __restrict__ res,
                              h16* __restrict__ Y,
                              int Cin, int Hin, int Win, int Cout, int Hout, int Wout,
                              int kh, int kw, int stride, int pad, int Kpad, int relu,
                              int Npix) {
    __shared__ __align__(16) h16 ldsA[2][64 * 64];   // 2 x 8 KB
    __shared__ __align__(16) h16 ldsB[2][64 * 64];   // 2 x 8 KB

    const int t    = threadIdx.x;
    const int lane = t & 31;
    const int wv   = t >> 5;
    const int wm   = wv & 1;          // M sub-pair: rows [wm*32, wm*32+32)
    const int wn   = wv >> 1;         // N sub-tile: cols [wn*16, wn*16+16)
    const int hh   = lane >> 4;
    const int lrow = lane & 15;

    const int n0 = blockIdx.x * 64;
    const int m0 = blockIdx.y * 64;

    // fill coords: row/col = t>>2 (0..63), seg = t&3 (16 halves = 2 x b128)
    const int frow = t >> 2, fseg = t & 3;
    const h16* wrowA = W + (long)(m0 + frow) * Kpad + fseg * 16;

    const int  nfill  = n0 + frow;
    const bool nvalid = nfill < Npix;
    int ox, oy, bb;
    {
        int nn = nvalid ? nfill : 0;
        ox = nn % Wout; int r = nn / Wout;
        oy = r % Hout;  bb = r / Hout;
    }
    const int iy0 = oy * stride - pad;
    const int ix0 = ox * stride - pad;
    const h16* Xb = X + (((long)bb * Hin + iy0) * Win + ix0) * Cin + fseg * 16;

    const int steps = kh * kw * (Cin >> 6);

    // "next tile to load" iterator state
    int ky = 0, kx = 0, ci0 = 0, ks = 0;

    auto load_tile = [&](int buf) {
        uint4 av0 = *reinterpret_cast<const uint4*>(wrowA + ks);
        uint4 av1 = *reinterpret_cast<const uint4*>(wrowA + ks + 8);
        __builtin_prefetch(wrowA + ks + 64, 0, 0);   // global_prefetch_b8
        uint4 bv0 = {0u, 0u, 0u, 0u}, bv1 = {0u, 0u, 0u, 0u};
        bool ok = nvalid &&
                  ((unsigned)(iy0 + ky) < (unsigned)Hin) &&
                  ((unsigned)(ix0 + kx) < (unsigned)Win);
        if (ok) {
            const h16* p = Xb + ((long)ky * Win + kx) * Cin + ci0;
            bv0 = *reinterpret_cast<const uint4*>(p);
            bv1 = *reinterpret_cast<const uint4*>(p + 8);
        }
        h16* da = &ldsA[buf][frow * 64 + fseg * 16];
        h16* db = &ldsB[buf][frow * 64 + fseg * 16];
        *reinterpret_cast<uint4*>(da)     = av0;
        *reinterpret_cast<uint4*>(da + 8) = av1;
        *reinterpret_cast<uint4*>(db)     = bv0;
        *reinterpret_cast<uint4*>(db + 8) = bv1;
    };
    auto advance = [&]() {
        ks += 64;
        ci0 += 64;
        if (ci0 == Cin) { ci0 = 0; ++kx; if (kx == kw) { kx = 0; ++ky; } }
    };

    v8f acc0 = {}, acc1 = {};

    load_tile(0);
    advance();
    __syncthreads();

    for (int s = 0; s < steps; ++s) {
        const int cur = s & 1;
        if (s + 1 < steps) {        // stage next tile into the other buffer
            load_tile(cur ^ 1);
            advance();
        }
        // 4 WMMAs from the current tile
        #pragma unroll
        for (int kk = 0; kk < 64; kk += 32) {
            V16 a0, a1, b;
            const h16* ap = &ldsA[cur][(wm * 32 + lrow) * 64 + kk + hh * 8];
            a0.q[0] = *reinterpret_cast<const uint4*>(ap);
            a0.q[1] = *reinterpret_cast<const uint4*>(ap + 16);
            a1.q[0] = *reinterpret_cast<const uint4*>(ap + 16 * 64);
            a1.q[1] = *reinterpret_cast<const uint4*>(ap + 16 * 64 + 16);
            const h16* bp = &ldsB[cur][(wn * 16 + lrow) * 64 + kk + hh * 8];
            b.q[0] = *reinterpret_cast<const uint4*>(bp);
            b.q[1] = *reinterpret_cast<const uint4*>(bp + 16);

            acc0 = __builtin_amdgcn_wmma_f32_16x16x32_f16(
                false, a0.v, false, b.v, (short)0, acc0, false, false);
            acc1 = __builtin_amdgcn_wmma_f32_16x16x32_f16(
                false, a1.v, false, b.v, (short)0, acc1, false, false);
        }
        __syncthreads();            // one barrier per K-step (double buffer)
    }

    const int nout = n0 + wn * 16 + lrow;
    if (nout < Npix) {
        #pragma unroll
        for (int r = 0; r < 8; ++r) {
            int co0 = m0 + wm * 32 + r + 8 * hh;
            int co1 = co0 + 16;
            long o0 = (long)nout * Cout + co0;
            long o1 = (long)nout * Cout + co1;
            float v0 = acc0[r] * sb[co0] + sb[Cout + co0];
            float v1 = acc1[r] * sb[co1] + sb[Cout + co1];
            if (res) { v0 += (float)res[o0]; v1 += (float)res[o1]; }
            if (relu) { v0 = fmaxf(v0, 0.f); v1 = fmaxf(v1, 0.f); }
            Y[o0] = (h16)v0;
            Y[o1] = (h16)v1;
        }
    }
}

// ---------------------------------------------------------------------------
// Generic path (conv1: Cin=3, 7x7).  Tile M=64 x N=32, scalar im2col gather.
// ---------------------------------------------------------------------------
__global__ __launch_bounds__(256)
void conv_wmma_kernel(const h16* __restrict__ X, const h16* __restrict__ W,
                      const float* __restrict__ sb, const h16* __restrict__ res,
                      h16* __restrict__ Y,
                      int Cin, int Hin, int Win, int Cout, int Hout, int Wout,
                      int kh, int kw, int stride, int pad, int Kpad, int relu) {
    __shared__ __align__(16) h16 ldsA[64 * 32];
    __shared__ __align__(16) h16 ldsB[32 * 32];

    const int t    = threadIdx.x;
    const int lane = t & 31;
    const int wv   = t >> 5;
    const int wm   = wv & 3;
    const int wn   = wv >> 2;
    const int hh   = lane >> 4;
    const int lrow = lane & 15;

    const int n0 = blockIdx.x * 32;
    const int m0 = blockIdx.y * 64;
    const int Ktot = Cin * kh * kw;

    const int ncol = t >> 3;
    const int kk0  = (t & 7) * 4;
    const int nfill = n0 + ncol;
    int ox = nfill % Wout; int tr = nfill / Wout;
    int oy = tr % Hout;    int bb = tr / Hout;

    const int arow = t >> 2, aseg = t & 3;
    const h16* wrow = W + (long)(m0 + arow) * Kpad + aseg * 8;

    v8f acc = {};

    for (int ks = 0; ks < Kpad; ks += 32) {
        uint4 av = *reinterpret_cast<const uint4*>(wrow + ks);
        *reinterpret_cast<uint4*>(&ldsA[arow * 32 + aseg * 8]) = av;

        #pragma unroll
        for (int j = 0; j < 4; ++j) {
            int k = ks + kk0 + j;
            h16 v = (h16)0.f;
            if (k < Ktot) {
                int ci = k % Cin; int r = k / Cin;
                int kx = r % kw;  int ky = r / kw;
                int iy = oy * stride - pad + ky;
                int ix = ox * stride - pad + kx;
                if ((unsigned)iy < (unsigned)Hin && (unsigned)ix < (unsigned)Win)
                    v = X[(((long)bb * Hin + iy) * Win + ix) * Cin + ci];
            }
            ldsB[ncol * 32 + kk0 + j] = v;
        }
        __syncthreads();

        V16 a, b;
        const h16* ap = &ldsA[(wm * 16 + lrow) * 32 + hh * 8];
        a.q[0] = *reinterpret_cast<const uint4*>(ap);
        a.q[1] = *reinterpret_cast<const uint4*>(ap + 16);
        const h16* bp = &ldsB[(wn * 16 + lrow) * 32 + hh * 8];
        b.q[0] = *reinterpret_cast<const uint4*>(bp);
        b.q[1] = *reinterpret_cast<const uint4*>(bp + 16);

        acc = __builtin_amdgcn_wmma_f32_16x16x32_f16(
            false, a.v, false, b.v, (short)0, acc, false, false);
        __syncthreads();
    }

    const int nout = n0 + wn * 16 + lrow;
    #pragma unroll
    for (int r = 0; r < 8; ++r) {
        int co = m0 + wm * 16 + r + 8 * hh;
        float v = acc[r] * sb[co] + sb[Cout + co];
        long oidx = (long)nout * Cout + co;
        if (res) v += (float)res[oidx];
        if (relu) v = fmaxf(v, 0.f);
        Y[oidx] = (h16)v;
    }
}

// ---------------------------------------------------------------------------
// 3x3/2 maxpool, pad 1, NHWC f16
// ---------------------------------------------------------------------------
__global__ void maxpool_kernel(const h16* __restrict__ X, h16* __restrict__ Y,
                               int C, int Hin, int Win, int Hout, int Wout) {
    long idx = (long)blockIdx.x * blockDim.x + threadIdx.x;
    long total = 32L * Hout * Wout * C;
    if (idx >= total) return;
    int c = (int)(idx % C); long r = idx / C;
    int ox = (int)(r % Wout); r /= Wout;
    int oy = (int)(r % Hout); int b = (int)(r / Hout);
    float m = -3.0e38f;
    #pragma unroll
    for (int ky = 0; ky < 3; ++ky)
        #pragma unroll
        for (int kx = 0; kx < 3; ++kx) {
            int iy = oy * 2 - 1 + ky, ix = ox * 2 - 1 + kx;
            if ((unsigned)iy < (unsigned)Hin && (unsigned)ix < (unsigned)Win)
                m = fmaxf(m, (float)X[(((long)b * Hin + iy) * Win + ix) * C + c]);
        }
    Y[idx] = (h16)m;
}

// global average pool: NHWC f16 [32][HW][C] -> f32 [32][C]
__global__ void avgpool_kernel(const h16* __restrict__ X, float* __restrict__ feat,
                               int C, int HW) {
    int idx = blockIdx.x * blockDim.x + threadIdx.x;
    if (idx >= 32 * C) return;
    int b = idx / C, c = idx % C;
    const h16* p = X + (long)b * HW * C + c;
    float acc = 0.f;
    for (int i = 0; i < HW; ++i) acc += (float)p[(long)i * C];
    feat[idx] = acc * (1.f / (float)HW);
}

// ---------------------------------------------------------------------------
// Head: fcr1+tanh -> fcr2 -> 4-qubit statevector sim -> final fc.  One block.
// ---------------------------------------------------------------------------
struct c64 { float x, y; };
__device__ __forceinline__ c64 cmul(c64 a, c64 b) { return {a.x*b.x - a.y*b.y, a.x*b.y + a.y*b.x}; }
__device__ __forceinline__ c64 cadd(c64 a, c64 b) { return {a.x + b.x, a.y + b.y}; }

__global__ __launch_bounds__(256)
void head_kernel(const float* __restrict__ feat,
                 const float* __restrict__ w1, const float* __restrict__ b1,
                 const float* __restrict__ w2, const float* __restrict__ b2,
                 const float* __restrict__ qw,
                 const float* __restrict__ fcw, const float* __restrict__ fcb,
                 float* __restrict__ out) {
    __shared__ float h[32 * 64];
    __shared__ float ang[32 * 4];
    const int t = threadIdx.x;

    for (int e = t; e < 32 * 64; e += 256) {
        int b = e >> 6, j = e & 63;
        float acc = b1[j];
        const float* f = feat + b * 512;
        const float* w = w1 + j * 512;
        for (int k = 0; k < 512; ++k) acc += f[k] * w[k];
        h[e] = tanhf(acc);
    }
    __syncthreads();

    if (t < 128) {
        int b = t >> 2, q = t & 3;
        float acc = b2[q];
        for (int k = 0; k < 64; ++k) acc += h[b * 64 + k] * w2[q * 64 + k];
        ang[t] = acc;
    }
    __syncthreads();

    if (t < 32) {
        c64 s[16];
        #pragma unroll
        for (int i = 0; i < 16; ++i) s[i] = {0.f, 0.f};
        s[0] = {1.f, 0.f};

        auto ap1 = [&](int w, c64 u00, c64 u01, c64 u10, c64 u11) {
            int bit = 8 >> w;  // wire 0 = MSB
            #pragma unroll
            for (int i = 0; i < 16; ++i)
                if (!(i & bit)) {
                    int j = i | bit;
                    c64 a = s[i], b = s[j];
                    s[i] = cadd(cmul(u00, a), cmul(u01, b));
                    s[j] = cadd(cmul(u10, a), cmul(u11, b));
                }
        };

        #pragma unroll
        for (int w = 0; w < 4; ++w) {   // RX embedding
            float a = ang[t * 4 + w];
            float ch = cosf(0.5f * a), sh = sinf(0.5f * a);
            ap1(w, {ch, 0.f}, {0.f, -sh}, {0.f, -sh}, {ch, 0.f});
        }
        for (int l = 0; l < 3; ++l) {   // StronglyEntanglingLayers
            #pragma unroll
            for (int w = 0; w < 4; ++w) {
                const float* q = qw + (l * 4 + w) * 3;
                float phi = q[0], th = q[1], om = q[2];
                float cb = cosf(0.5f * th), sbv = sinf(0.5f * th);
                float xs = 0.5f * (phi + om), ys = 0.5f * (phi - om);
                c64 u00 = { cosf(xs) * cb, -sinf(xs) * cb };
                c64 u01 = { -cosf(ys) * sbv, -sinf(ys) * sbv };
                c64 u10 = { cosf(ys) * sbv, -sinf(ys) * sbv };
                c64 u11 = { cosf(xs) * cb,  sinf(xs) * cb };
                ap1(w, u00, u01, u10, u11);
            }
            int rr = (l % 3) + 1;
            #pragma unroll
            for (int w = 0; w < 4; ++w) {   // CNOT ring
                int bc = 8 >> w, bt = 8 >> ((w + rr) & 3);
                #pragma unroll
                for (int i = 0; i < 16; ++i)
                    if ((i & bc) && !(i & bt)) {
                        int j = i | bt;
                        c64 tmp = s[i]; s[i] = s[j]; s[j] = tmp;
                    }
            }
        }
        float z[4];
        #pragma unroll
        for (int w = 0; w < 4; ++w) {
            int bit = 8 >> w; float acc = 0.f;
            #pragma unroll
            for (int i = 0; i < 16; ++i) {
                float p = s[i].x * s[i].x + s[i].y * s[i].y;
                acc += (i & bit) ? -p : p;
            }
            z[w] = acc;
        }
        #pragma unroll
        for (int o = 0; o < 2; ++o) {
            float v = fcb[o];
            #pragma unroll
            for (int w = 0; w < 4; ++w) v += z[w] * fcw[o * 4 + w];
            out[t * 2 + o] = v;
        }
    }
}

// ---------------------------------------------------------------------------
// host orchestration
// ---------------------------------------------------------------------------
extern "C" void kernel_launch(void* const* d_in, const int* in_sizes, int n_in,
                              void* d_out, int out_size, void* d_ws, size_t ws_size,
                              hipStream_t stream) {
    (void)in_sizes; (void)n_in; (void)out_size; (void)ws_size;

    char* wsp = (char*)d_ws;
    size_t off = 0;
    auto alloc = [&](size_t bytes) -> void* {
        off = (off + 255) & ~(size_t)255;
        void* p = wsp + off;
        off += bytes;
        return p;
    };

    h16* x16 = (h16*)alloc(32L * 224 * 224 * 3 * sizeof(h16));
    h16* a1  = (h16*)alloc(32L * 112 * 112 * 64 * sizeof(h16));
    size_t actBytes = 32L * 56 * 56 * 64 * sizeof(h16);
    h16* bufs[4];
    for (int i = 0; i < 4; ++i) bufs[i] = (h16*)alloc(actBytes);
    float* feat = (float*)alloc(32L * 512 * sizeof(float));

    auto run_conv = [&](int widx, const h16* X, const h16* res, h16* Y,
                        int Cin, int Hin, int Cout, int k, int stride, int pad,
                        int relu) -> int {
        int Hout = (Hin + 2 * pad - k) / stride + 1;
        int Ktot = Cin * k * k;
        int Kpad = (Ktot + 31) & ~31;
        h16*   W16 = (h16*)alloc((size_t)Cout * Kpad * sizeof(h16));
        float* sb  = (float*)alloc((size_t)2 * Cout * sizeof(float));
        int np = Cout * Kpad;
        prep_w_kernel<<<(np + 255) / 256, 256, 0, stream>>>(
            (const float*)d_in[widx], W16, Cout, Cin, k, k, Kpad);
        prep_bn_kernel<<<(Cout + 255) / 256, 256, 0, stream>>>(
            (const float*)d_in[widx + 1], (const float*)d_in[widx + 2],
            (const float*)d_in[widx + 3], (const float*)d_in[widx + 4], sb, Cout);
        int Npix = 32 * Hout * Hout;
        if ((Cin & 63) == 0) {
            dim3 grid((Npix + 63) / 64, Cout / 64);
            conv_wmma_aligned_kernel<<<grid, 256, 0, stream>>>(
                X, W16, sb, res, Y, Cin, Hin, Hin, Cout, Hout, Hout,
                k, k, stride, pad, Kpad, relu, Npix);
        } else {
            dim3 grid(Npix / 32, Cout / 64);
            conv_wmma_kernel<<<grid, 256, 0, stream>>>(
                X, W16, sb, res, Y, Cin, Hin, Hin, Cout, Hout, Hout,
                k, k, stride, pad, Kpad, relu);
        }
        return Hout;
    };

    {
        long n = 32L * 224 * 224 * 3;
        nchw_to_nhwc_kernel<<<(int)((n + 255) / 256), 256, 0, stream>>>(
            (const float*)d_in[0], x16, 3, 224, 224);
    }
    run_conv(1, x16, nullptr, a1, 3, 224, 64, 7, 2, 3, 1);
    {
        long n = 32L * 56 * 56 * 64;
        maxpool_kernel<<<(int)((n + 255) / 256), 256, 0, stream>>>(
            a1, bufs[0], 64, 112, 112, 56, 56);
    }

    int curi = 0, H = 56, widx = 6;
    auto block = [&](int Cin, int Cout, int stride) {
        int t1 = (curi + 1) & 3, t2 = (curi + 2) & 3, t3 = (curi + 3) & 3;
        int Hout = run_conv(widx, bufs[curi], nullptr, bufs[t1],
                            Cin, H, Cout, 3, stride, 1, 1);
        const h16* res = bufs[curi];
        if (stride == 2) {
            run_conv(widx + 10, bufs[curi], nullptr, bufs[t2],
                     Cin, H, Cout, 1, stride, 0, 0);
            res = bufs[t2];
        }
        run_conv(widx + 5, bufs[t1], res, bufs[t3], Cout, Hout, Cout, 3, 1, 1, 1);
        widx += (stride == 2) ? 15 : 10;
        curi = t3;
        H = Hout;
    };

    block(64, 64, 1);   block(64, 64, 1);      // layer1
    block(64, 128, 2);  block(128, 128, 1);    // layer2
    block(128, 256, 2); block(256, 256, 1);    // layer3
    block(256, 512, 2); block(512, 512, 1);    // layer4

    avgpool_kernel<<<(32 * 512 + 255) / 256, 256, 0, stream>>>(
        bufs[curi], feat, 512, 49);

    head_kernel<<<1, 256, 0, stream>>>(
        feat,
        (const float*)d_in[101], (const float*)d_in[102],
        (const float*)d_in[103], (const float*)d_in[104],
        (const float*)d_in[105],
        (const float*)d_in[106], (const float*)d_in[107],
        (float*)d_out);
}
